// ProteinEncoder_65489661329901
// MI455X (gfx1250) — matrix operand order repaired and checked by reading, not compiled, not run on hardware
//
#include <hip/hip_runtime.h>
#include <stdint.h>

#define NN      16384      // nodes
#define HH      128        // hidden
#define LL      3          // layers
#define KK      32         // neighbors
#define NRBF    16
#define VOCAB   21
#define NE      (NN * KK)  // edges
#define SEP     132        // padded e-tile row (floats): 528B = 33*16B, bank-conflict-free

typedef __attribute__((ext_vector_type(16))) __bf16 v16bf;
typedef __attribute__((ext_vector_type(8)))  float  v8f;

union BF16Frag { v16bf v; unsigned short s[16]; };

static __device__ __forceinline__ unsigned short f2bf(float x) {
    // round-to-nearest-even f32 -> bf16
    unsigned int u = __float_as_uint(x);
    unsigned int r = u + 0x7FFFu + ((u >> 16) & 1u);
    return (unsigned short)(r >> 16);
}

// ---------------------------------------------------------------------------
// 0) T[v][h] = emb[v] @ node_W + node_b   (21x128, folds node GEMM into gather)
// ---------------------------------------------------------------------------
__global__ void node_table_kernel(const float* __restrict__ emb,
                                  const float* __restrict__ node_W,
                                  const float* __restrict__ node_b,
                                  float* __restrict__ T) {
    for (int o = threadIdx.x; o < VOCAB * HH; o += blockDim.x) {
        int v = o >> 7, h = o & 127;
        float acc = node_b[h];
        for (int d = 0; d < 64; ++d)
            acc = fmaf(emb[v * 64 + d], node_W[d * HH + h], acc);
        T[o] = acc;
    }
}

// ---------------------------------------------------------------------------
// 1) h0[i] = T[res_type[i]]
// ---------------------------------------------------------------------------
__global__ void h_init_kernel(const int* __restrict__ res,
                              const float* __restrict__ T,
                              float* __restrict__ h) {
    int i = blockIdx.x * blockDim.x + threadIdx.x;
    if (i < NN * HH) {
        int n = i >> 7, c = i & 127;
        h[i] = T[res[n] * HH + c];
    }
}

// ---------------------------------------------------------------------------
// 2) brute-force KNN, top-32 per query, LDS-tiled candidates,
//    per-thread sorted list kept in LDS (ascending d2).
// ---------------------------------------------------------------------------
#define KNN_T 128
__global__ void knn_kernel(const float* __restrict__ pos,
                           int* __restrict__ idx_out,
                           float* __restrict__ dist_out) {
    __shared__ float sx[256], sy[256], sz[256];
    __shared__ float sdist[KK][KNN_T];
    __shared__ int   sidx[KK][KNN_T];

    int t = threadIdx.x;
    int i = blockIdx.x * KNN_T + t;
    float qx = pos[i * 3 + 0], qy = pos[i * 3 + 1], qz = pos[i * 3 + 2];

    #pragma unroll
    for (int k = 0; k < KK; ++k) { sdist[k][t] = 3.0e38f; sidx[k][t] = 0; }
    float worst = 3.0e38f;

    for (int c0 = 0; c0 < NN; c0 += 256) {
        __syncthreads();
        for (int b = t; b < 256; b += KNN_T) {
            int j = c0 + b;
            sx[b] = pos[j * 3 + 0]; sy[b] = pos[j * 3 + 1]; sz[b] = pos[j * 3 + 2];
        }
        __syncthreads();
        for (int b = 0; b < 256; ++b) {
            int j = c0 + b;
            float dx = qx - sx[b], dy = qy - sy[b], dz = qz - sz[b];
            float d2 = fmaf(dx, dx, fmaf(dy, dy, dz * dz));
            if (j != i && d2 < worst) {
                int p = KK - 1;
                while (p > 0 && sdist[p - 1][t] > d2) {
                    sdist[p][t] = sdist[p - 1][t];
                    sidx[p][t]  = sidx[p - 1][t];
                    --p;
                }
                sdist[p][t] = d2;
                sidx[p][t]  = j;
                worst = sdist[KK - 1][t];
            }
        }
    }
    #pragma unroll
    for (int k = 0; k < KK; ++k) {
        idx_out[i * KK + k]  = sidx[k][t];
        dist_out[i * KK + k] = sqrtf(sdist[k][t]);
    }
}

// ---------------------------------------------------------------------------
// 3) e0[edge] = rbf(dist) @ edge_W + edge_b   (16 edges / 256-thread block)
// ---------------------------------------------------------------------------
__global__ void edge_init_kernel(const float* __restrict__ dist,
                                 const float* __restrict__ edge_W,
                                 const float* __restrict__ edge_b,
                                 float* __restrict__ e) {
    __shared__ float sW[NRBF][HH];
    __shared__ float sb[HH];
    __shared__ float srbf[16][NRBF];
    int t = threadIdx.x;
    for (int i = t; i < NRBF * HH; i += 256) sW[i >> 7][i & 127] = edge_W[i];
    if (t < HH) sb[t] = edge_b[t];

    int e0 = blockIdx.x * 16;
    {
        int el = t >> 4, c = t & 15;
        float d = dist[e0 + el];
        float center = (8.0f / 15.0f) * (float)c;
        float z = (d - center) * 2.0f;        // gamma = 0.5
        srbf[el][c] = __expf(-z * z);
    }
    __syncthreads();

    int el = t >> 4;
    int h0 = (t & 15) * 8;
    float acc[8];
    #pragma unroll
    for (int r = 0; r < 8; ++r) acc[r] = sb[h0 + r];
    #pragma unroll
    for (int c = 0; c < NRBF; ++c) {
        float rb = srbf[el][c];
        #pragma unroll
        for (int r = 0; r < 8; ++r) acc[r] = fmaf(rb, sW[c][h0 + r], acc[r]);
    }
    long base = (long)(e0 + el) * HH + h0;
    #pragma unroll
    for (int r = 0; r < 8; ++r) e[base + r] = acc[r];
}

// ---------------------------------------------------------------------------
// 4) out[M,128] = X[M,128] @ W[128,128] + b   via v_wmma_f32_16x16x32_bf16
//    (used for the four h-side GEMMs: A,B,V,U)
// ---------------------------------------------------------------------------
__global__ void gemm128_kernel(const float* __restrict__ X,
                               const float* __restrict__ W,
                               const float* __restrict__ bias,
                               float* __restrict__ out, int M) {
    __shared__ __align__(32) unsigned short Wlds[4][8][32][16];

    int t = threadIdx.x;
    for (int i = t; i < HH * HH; i += 256) {
        int k = i >> 7, n = i & 127;
        int kt = k >> 5, kk = k & 31, nt = n >> 4, nn = n & 15;
        int lane = nn + (kk & 16);
        int pos  = kk & 15;
        Wlds[kt][nt][lane][pos] = f2bf(W[i]);
    }
    __syncthreads();

    int wave = t >> 5, lane = t & 31;
    long m0 = ((long)blockIdx.x * 8 + wave) * 16;
    if (m0 >= M) return;

    int arow  = lane & 15;
    int khalf = (lane >> 4) ? 16 : 0;
    const float* xrow = X + (m0 + arow) * HH;

    BF16Frag af[4];
    #pragma unroll
    for (int k4 = 0; k4 < 4; ++k4) {
        const float* p0 = xrow + k4 * 32 + khalf;
        const float* p1 = xrow + k4 * 32 + khalf + 8;
        #pragma unroll
        for (int j = 0; j < 8; ++j) {
            af[k4].s[j]     = f2bf(p0[j]);
            af[k4].s[8 + j] = f2bf(p1[j]);
        }
    }

    #pragma unroll
    for (int nt = 0; nt < 8; ++nt) {
        float bv = bias[nt * 16 + (lane & 15)];
        v8f acc;
        #pragma unroll
        for (int r = 0; r < 8; ++r) acc[r] = bv;

        #pragma unroll
        for (int k4 = 0; k4 < 4; ++k4) {
            v16bf bfrag = *(const v16bf*)&Wlds[k4][nt][lane][0];
            acc = __builtin_amdgcn_wmma_f32_16x16x32_bf16(
                false, af[k4].v, false, bfrag, (short)0, acc, false, false);
        }
        float* orow = out + m0 * HH + nt * 16 + (lane & 15);
        int mo = (lane >> 4) * 8;
        #pragma unroll
        for (int r = 0; r < 8; ++r) orow[(long)(mo + r) * HH] = acc[r];
    }
}

// ---------------------------------------------------------------------------
// 5) pack CW[l] (f32 row-major) into B-fragment-layout bf16 (one block)
// ---------------------------------------------------------------------------
__global__ void pack_cw_kernel(const float* __restrict__ W,
                               unsigned short* __restrict__ Wp) {
    for (int i = threadIdx.x; i < HH * HH; i += blockDim.x) {
        int k = i >> 7, n = i & 127;
        int kt = k >> 5, kk = k & 31, nt = n >> 4, nn = n & 15;
        int lane = nn + (kk & 16);
        int pos  = kk & 15;
        Wp[((kt * 8 + nt) * 32 + lane) * 16 + pos] = f2bf(W[i]);
    }
}

// ---------------------------------------------------------------------------
// 6) fused edge GEMM + gather/gate/aggregate/update.
//    One 128-thread block (4 waves) per node; e-tile lives in LDS.
//    wave w: m-tile = w&1 (k rows), n-tiles = (w>>1)*4 .. +3 (channels)
// ---------------------------------------------------------------------------
__global__ void edge_fuse_kernel(const float* __restrict__ h,
                                 const float* __restrict__ Ah,
                                 const float* __restrict__ Bh,
                                 const float* __restrict__ Vh,
                                 const float* __restrict__ Uh,
                                 const float* __restrict__ Cb,
                                 const unsigned short* __restrict__ CWp,
                                 float* __restrict__ e,        // updated in place
                                 const int* __restrict__ idx,
                                 const float* __restrict__ dist,
                                 float* __restrict__ h_out) {
    __shared__ __align__(16) float se[KK][SEP];               // padded e tile
    __shared__ __align__(32) unsigned short cw[4][8][32][16]; // B fragments
    __shared__ float snum[4][HH], sden[4][HH];
    __shared__ float sAhCb[HH];
    __shared__ int   sj[KK];
    __shared__ float sv[KK];

    int i = blockIdx.x;
    int t = threadIdx.x;          // 0..127
    int wave = t >> 5, lane = t & 31;

    // ---- phase 1: stage e tile, packed CW, per-node vectors ----
    {
        const uint4* esrc = (const uint4*)(e + (long)i * KK * HH);
        for (int g4 = t; g4 < (KK * HH) / 4; g4 += 128) {
            int k = g4 >> 5, c4 = (g4 & 31) * 4;
            *(uint4*)&se[k][c4] = esrc[g4];
        }
        const uint4* wsrc = (const uint4*)CWp;
        uint4* wdst = (uint4*)&cw[0][0][0][0];
        for (int g4 = t; g4 < (4 * 8 * 32 * 16) / 8; g4 += 128)
            wdst[g4] = wsrc[g4];
        if (t < HH) sAhCb[t] = Ah[(long)i * HH + t] + Cb[t];
        if (t < KK) {
            sj[t] = idx[i * KK + t];
            sv[t] = (dist[i * KK + t] <= 8.0f) ? 1.0f : 0.0f;
        }
    }
    __syncthreads();

    // ---- phase 2: build A fragments (bf16) for this wave's m-tile ----
    int mt = wave & 1;
    int ng = (wave >> 1) * 4;
    int arow  = mt * 16 + (lane & 15);
    int khoff = (lane >> 4) ? 16 : 0;
    BF16Frag af[4];
    #pragma unroll
    for (int k4 = 0; k4 < 4; ++k4) {
        const float* p0 = &se[arow][k4 * 32 + khoff];
        const float* p1 = &se[arow][k4 * 32 + khoff + 8];
        #pragma unroll
        for (int j = 0; j < 8; ++j) {
            af[k4].s[j]     = f2bf(p0[j]);
            af[k4].s[8 + j] = f2bf(p1[j]);
        }
    }
    __syncthreads();   // every wave finished reading se before e_new overwrites it

    // ---- phase 3: WMMA + elementwise gating on accumulators ----
    int colbase = lane & 15;
    int part  = 2 * mt + (lane >> 4);
    int khalf = (lane >> 4) * 8;
    #pragma unroll
    for (int q = 0; q < 4; ++q) {
        int nt  = ng + q;
        int col = nt * 16 + colbase;
        float base = sAhCb[col];
        v8f acc;
        #pragma unroll
        for (int r = 0; r < 8; ++r) acc[r] = base;

        #pragma unroll
        for (int k4 = 0; k4 < 4; ++k4) {
            v16bf bfrag = *(const v16bf*)&cw[k4][nt][lane][0];
            acc = __builtin_amdgcn_wmma_f32_16x16x32_bf16(
                false, af[k4].v, false, bfrag, (short)0, acc, false, false);
        }

        float numpart = 0.0f, denpart = 0.0f;
        #pragma unroll
        for (int r = 0; r < 8; ++r) {
            int k = mt * 16 + khalf + r;          // D layout: M = r + 8*(lane/16)
            int j = sj[k];
            float ehat = acc[r] + Bh[(long)j * HH + col];
            float sg   = sv[k] / (1.0f + __expf(-ehat));
            numpart = fmaf(sg, Vh[(long)j * HH + col], numpart);
            denpart += sg;
            se[k][col] += fmaxf(ehat, 0.0f);      // e_new (own region only)
        }
        snum[part][col] = numpart;                // unique writer per slot
        sden[part][col] = denpart;
    }
    __syncthreads();

    // ---- phase 4: coalesced e writeback + h update ----
    {
        float* edst = e + (long)i * KK * HH;
        for (int g4 = t; g4 < (KK * HH) / 4; g4 += 128) {
            int k = g4 >> 5, c4 = (g4 & 31) * 4;
            *(uint4*)(edst + g4 * 4) = *(const uint4*)&se[k][c4];
        }
        if (t < HH) {
            float num = snum[0][t] + snum[1][t] + snum[2][t] + snum[3][t];
            float den = sden[0][t] + sden[1][t] + sden[2][t] + sden[3][t];
            float upd = Uh[(long)i * HH + t] + num / (den + 1e-6f);
            h_out[(long)i * HH + t] = h[(long)i * HH + t] + fmaxf(upd, 0.0f);
        }
    }
}

// ---------------------------------------------------------------------------
extern "C" void kernel_launch(void* const* d_in, const int* in_sizes, int n_in,
                              void* d_out, int out_size, void* d_ws, size_t ws_size,
                              hipStream_t stream) {
    const float* pos      = (const float*)d_in[0];
    const int*   res_type = (const int*)  d_in[1];
    const float* emb      = (const float*)d_in[2];
    const float* node_W   = (const float*)d_in[3];
    const float* node_b   = (const float*)d_in[4];
    const float* edge_W   = (const float*)d_in[5];
    const float* edge_b   = (const float*)d_in[6];
    const float* AW = (const float*)d_in[7];
    const float* Ab = (const float*)d_in[8];
    const float* BW = (const float*)d_in[9];
    const float* Bb = (const float*)d_in[10];
    const float* CW = (const float*)d_in[11];
    const float* Cb = (const float*)d_in[12];
    const float* UW = (const float*)d_in[13];
    const float* Ub = (const float*)d_in[14];
    const float* VW = (const float*)d_in[15];
    const float* Vb = (const float*)d_in[16];

    char* ws = (char*)d_ws;
    size_t off = 0;
    auto alloc = [&](size_t bytes) -> void* {
        void* p = ws + off;
        off += (bytes + 255) & ~(size_t)255;
        return p;
    };
    float* T    = (float*)alloc((size_t)VOCAB * HH * 4);
    int*   idx  = (int*)  alloc((size_t)NE * 4);
    float* dist = (float*)alloc((size_t)NE * 4);
    float* h    = (float*)alloc((size_t)NN * HH * 4);
    float* Ah   = (float*)alloc((size_t)NN * HH * 4);
    float* Bh   = (float*)alloc((size_t)NN * HH * 4);
    float* Vhb  = (float*)alloc((size_t)NN * HH * 4);
    float* Uhb  = (float*)alloc((size_t)NN * HH * 4);
    unsigned short* CWp = (unsigned short*)alloc((size_t)4 * 8 * 32 * 16 * 2);
    float* e    = (float*)alloc((size_t)NE * HH * 4);

    hipLaunchKernelGGL(node_table_kernel, dim3(1), dim3(256), 0, stream,
                       emb, node_W, node_b, T);
    hipLaunchKernelGGL(h_init_kernel, dim3((NN * HH) / 256), dim3(256), 0, stream,
                       res_type, T, h);
    hipLaunchKernelGGL(knn_kernel, dim3(NN / KNN_T), dim3(KNN_T), 0, stream,
                       pos, idx, dist);
    hipLaunchKernelGGL(edge_init_kernel, dim3(NE / 16), dim3(256), 0, stream,
                       dist, edge_W, edge_b, e);

    const int gemmN_blocks = NN / (16 * 8);   // 128

    for (int l = 0; l < LL; ++l) {
        const float* aw = AW + (size_t)l * HH * HH;
        const float* bw = BW + (size_t)l * HH * HH;
        const float* cw = CW + (size_t)l * HH * HH;
        const float* uw = UW + (size_t)l * HH * HH;
        const float* vw = VW + (size_t)l * HH * HH;
        const float* ab = Ab + (size_t)l * HH;
        const float* bb = Bb + (size_t)l * HH;
        const float* cb = Cb + (size_t)l * HH;
        const float* ub = Ub + (size_t)l * HH;
        const float* vb = Vb + (size_t)l * HH;

        hipLaunchKernelGGL(pack_cw_kernel, dim3(1), dim3(256), 0, stream, cw, CWp);

        hipLaunchKernelGGL(gemm128_kernel, dim3(gemmN_blocks), dim3(256), 0, stream,
                           h, aw, ab, Ah, NN);
        hipLaunchKernelGGL(gemm128_kernel, dim3(gemmN_blocks), dim3(256), 0, stream,
                           h, bw, bb, Bh, NN);
        hipLaunchKernelGGL(gemm128_kernel, dim3(gemmN_blocks), dim3(256), 0, stream,
                           h, vw, vb, Vhb, NN);
        hipLaunchKernelGGL(gemm128_kernel, dim3(gemmN_blocks), dim3(256), 0, stream,
                           h, uw, ub, Uhb, NN);

        float* hout = (l == LL - 1) ? (float*)d_out : h;
        hipLaunchKernelGGL(edge_fuse_kernel, dim3(NN), dim3(HH), 0, stream,
                           h, Ah, Bh, Vhb, Uhb, cb, CWp, e, idx, dist, hout);
    }
}
